// GCNEncoder_4896262717965
// MI455X (gfx1250) — matrix operand order repaired and checked by reading, not compiled, not run on hardware
//
#include <hip/hip_runtime.h>
#include <float.h>

// ---------------------------------------------------------------------------
// DGCNN forward for MI455X (gfx1250, wave32, WMMA f32<=f16 16x16x32)
// B=16 clouds, N=4096 pts, K=20 neighbors.
// ---------------------------------------------------------------------------

#define BN_SCALE 0.99999500003749977f   // 1/sqrt(1+1e-5), eval-mode BatchNorm
#define NPTS     4096
#define NTOT     (16 * 4096)
#define KNN      20

typedef __attribute__((ext_vector_type(16))) _Float16 v16h;
typedef __attribute__((ext_vector_type(8)))  _Float16 v8h;
typedef __attribute__((ext_vector_type(8)))  float    v8f;

union frag_u {
  v16h v;
  v8h  h[2];
};

// ---------------- WMMA helpers (CDNA5 layouts per ISA 7.12.2) ----------------

__device__ __forceinline__ v8f wmma16(v16h a, v16h b, v8f c) {
  // D = A(16x32 f16) x B(32x16 f16) + C(16x16 f32)
  return __builtin_amdgcn_wmma_f32_16x16x32_f16(false, a, false, b, (short)0, c,
                                                false, false);
}

// A fragment: row-major LDS tile [16][strideH] halves, K-window [k0,k0+32).
// Lane L: M = L&15, K = k0 + 8*(L>>4) + (e<8 ? e : e+8)  -> two 16B ds loads.
__device__ __forceinline__ v16h load_a_frag(const _Float16* base, int strideH,
                                            int k0, int lane) {
  int m = lane & 15;
  int kofs = (lane >> 4) * 8;
  const _Float16* p = base + m * strideH + k0 + kofs;
  frag_u u;
  u.h[0] = *(const v8h*)(p);
  u.h[1] = *(const v8h*)(p + 16);
  return u.v;
}

// B fragment from pre-swizzled LDS: [tile][lane][16 halves], 32B contiguous.
__device__ __forceinline__ v16h load_b_frag_sw(const _Float16* Wfrag, int tIdx,
                                               int lane) {
  const _Float16* p = Wfrag + tIdx * 512 + lane * 16;
  frag_u u;
  u.h[0] = *(const v8h*)(p);
  u.h[1] = *(const v8h*)(p + 8);
  return u.v;
}

// Monotone float <-> uint key for atomicMax-based float max.
__device__ __forceinline__ unsigned f2key(float f) {
  unsigned u = __float_as_uint(f);
  return (u & 0x80000000u) ? ~u : (u | 0x80000000u);
}
__device__ __forceinline__ float key2f(unsigned k) {
  unsigned u = (k & 0x80000000u) ? (k & 0x7FFFFFFFu) : ~k;
  return __uint_as_float(u);
}

// Swizzled-weight staging index decode:
//   i = tIdx*512 + ln*16 + e  ->  (k, n) with
//   k = ks*32 + (ln>>4)*8 + (e&7) + (e>>3)*16,  n = nt*16 + (ln&15),  tIdx = ks*4 + nt
__device__ __forceinline__ void sw_decode(int i, int& k, int& n) {
  int e = i & 15;
  int ln = (i >> 4) & 31;
  int tIdx = i >> 9;
  int ks = tIdx >> 2, nt = tIdx & 3;
  k = ks * 32 + (ln >> 4) * 8 + (e & 7) + (e >> 3) * 16;
  n = nt * 16 + (ln & 15);
}

// ---------------- squared norms ----------------

__global__ __launch_bounds__(256) void norm_kernel(const float* __restrict__ x,
                                                   int dim,
                                                   float* __restrict__ xn) {
  int p = blockIdx.x * 256 + threadIdx.x;
  if (p >= NTOT) return;
  float s = 0.f;
  for (int c = 0; c < dim; ++c) { float v = x[(size_t)p * dim + c]; s += v * v; }
  xn[p] = s;
}

// ---------------- kNN: one workgroup (256 thr) per query point --------------
// Score s_j = |x_j|^2 - 2 x_q . x_j  (same ordering as full distance).
// 20 rounds of masked argmin, ties -> lowest index (matches lax.top_k).

__global__ __launch_bounds__(256) void knn_kernel(const float* __restrict__ x,
                                                  const float* __restrict__ xn,
                                                  int dim,
                                                  int* __restrict__ knn) {
  __shared__ float xq[64];
  __shared__ float sd[NPTS];
  __shared__ float rv[8];
  __shared__ int   ri[8];
  __shared__ int   out_l[KNN];

  int q = blockIdx.x;
  int base = (q >> 12) << 12;  // batch start (4096 per batch)
  int t = threadIdx.x;

  for (int c = t; c < dim; c += 256) xq[c] = x[(size_t)q * dim + c];
  __syncthreads();

  for (int jj = 0; jj < 16; ++jj) {
    int j = t + jj * 256;
    const float* xp = x + (size_t)(base + j) * dim;
    float s = xn[base + j];
    for (int c = 0; c < dim; ++c) s -= 2.f * xq[c] * xp[c];
    sd[j] = s;
  }
  __syncthreads();

  int lane = t & 31, w = t >> 5;
  for (int r = 0; r < KNN; ++r) {
    float mv = FLT_MAX;
    int mi = -1;
    for (int jj = 0; jj < 16; ++jj) {
      int j = t + jj * 256;
      float v = sd[j];
      if (v < mv || (v == mv && (unsigned)j < (unsigned)mi)) { mv = v; mi = j; }
    }
#pragma unroll
    for (int off = 16; off > 0; off >>= 1) {
      float ov = __shfl_down(mv, off);
      int   oi = __shfl_down(mi, off);
      if (ov < mv || (ov == mv && (unsigned)oi < (unsigned)mi)) { mv = ov; mi = oi; }
    }
    if (lane == 0) { rv[w] = mv; ri[w] = mi; }
    __syncthreads();
    if (t == 0) {
      float bm = rv[0]; int bi = ri[0];
      for (int ww = 1; ww < 8; ++ww)
        if (rv[ww] < bm || (rv[ww] == bm && (unsigned)ri[ww] < (unsigned)bi)) {
          bm = rv[ww]; bi = ri[ww];
        }
      out_l[r] = bi;
      sd[bi] = FLT_MAX;  // mask
    }
    __syncthreads();
  }
  if (t < KNN) knn[(size_t)q * KNN + t] = base + out_l[t];
}

// ---------------- fused EdgeConv MLP + max over neighbors -------------------
// Block = 128 thr (4 waves) handles 16 points. Each wave processes neighbor
// slots k = w, w+4, ..., w+16 (5 tiles). Per tile: build A=[x_i, x_j-x_i]
// (16 x KD1 f16) in LDS, h1 = WMMA GEMM + bias/ReLU/BN, h2 likewise, then
// elementwise running max in registers; cross-wave max via LDS at the end.
// Weights live in LDS pre-swizzled to WMMA-fragment order (2x ds_load_b128
// per fragment); layer-2 fragments are additionally hoisted into registers.

template <int KD1, int DIN>
__global__ __launch_bounds__(128) void edgeconv_kernel(
    const float* __restrict__ x, const int* __restrict__ knn,
    const float* __restrict__ W1, const float* __restrict__ b1,
    const float* __restrict__ g1, const float* __restrict__ t1,
    const float* __restrict__ W2, const float* __restrict__ b2,
    const float* __restrict__ g2, const float* __restrict__ t2,
    float* __restrict__ y) {
  constexpr int NT1 = (KD1 / 32) * 4;  // layer-1 fragment tiles
  constexpr int ABYTES = 4 * 16 * KD1 * 2;
  constexpr int HBYTES = 4 * 16 * 64 * 2;
  constexpr int MBYTES = 4 * 16 * 64 * 4;
  constexpr int UBYTES = (ABYTES + HBYTES > MBYTES) ? (ABYTES + HBYTES) : MBYTES;

  __shared__ __align__(16) _Float16 Wf1[NT1 * 512];  // swizzled fragments
  __shared__ __align__(16) _Float16 Wf2[8 * 512];
  __shared__ __align__(16) char smem[UBYTES];
  _Float16* Astage = (_Float16*)smem;
  _Float16* Hstage = (_Float16*)(smem + ABYTES);
  float*    maxbuf = (float*)smem;  // overlays A/H after final barrier

  int tid = threadIdx.x;
  int lane = tid & 31, w = tid >> 5;
  int nlo = lane & 15, half = lane >> 4;
  int p0 = blockIdx.x * 16;

  // Stage weights (f32 -> f16) in fragment order; zero-pad rows >= 2*DIN.
  for (int i = tid; i < NT1 * 512; i += 128) {
    int k, n;
    sw_decode(i, k, n);
    Wf1[i] = (_Float16)((k < 2 * DIN) ? W1[k * 64 + n] : 0.f);
  }
  for (int i = tid; i < 8 * 512; i += 128) {
    int k, n;
    sw_decode(i, k, n);
    Wf2[i] = (_Float16)W2[k * 64 + n];
  }

  // Per-lane epilogue constants.
  float b1c[4], g1c[4], t1c[4], b2c[4], g2c[4], t2c[4];
#pragma unroll
  for (int nt = 0; nt < 4; ++nt) {
    int c = nt * 16 + nlo;
    b1c[nt] = b1[c]; g1c[nt] = g1[c] * BN_SCALE; t1c[nt] = t1[c];
    b2c[nt] = b2[c]; g2c[nt] = g2[c] * BN_SCALE; t2c[nt] = t2[c];
  }
  __syncthreads();

  // Hoist layer-2 weight fragments into registers (reused by all 5 tiles).
  v16h w2f[8];
#pragma unroll
  for (int f = 0; f < 8; ++f) w2f[f] = load_b_frag_sw(Wf2, f, lane);

  _Float16* Aw = Astage + w * 16 * KD1;
  _Float16* Hw = Hstage + w * 16 * 64;

  float vmax[4][8];
#pragma unroll
  for (int nt = 0; nt < 4; ++nt)
#pragma unroll
    for (int i = 0; i < 8; ++i) vmax[nt][i] = -FLT_MAX;

  for (int kk = w; kk < KNN; kk += 4) {
    // --- stage A: rows = 16 points, cols = [x_i | x_j - x_i | zero-pad] ----
    {
      int m = nlo;
      int p = p0 + m;
      int nb = knn[(size_t)p * KNN + kk];
#pragma unroll
      for (int c0 = 0; c0 < KD1 / 2; ++c0) {
        int c = half * (KD1 / 2) + c0;
        float v;
        if (c < DIN)          v = x[(size_t)p * DIN + c];
        else if (c < 2 * DIN) v = x[(size_t)nb * DIN + (c - DIN)] -
                                  x[(size_t)p * DIN + (c - DIN)];
        else                  v = 0.f;
        Aw[m * KD1 + c] = (_Float16)v;
      }
    }

    // --- layer 1: 16 x KD1 @ KD1 x 64 ---
#pragma unroll
    for (int nt = 0; nt < 4; ++nt) {
      v8f acc;
#pragma unroll
      for (int i = 0; i < 8; ++i) acc[i] = 0.f;
#pragma unroll
      for (int ks = 0; ks < KD1 / 32; ++ks) {
        v16h a = load_a_frag(Aw, KD1, ks * 32, lane);
        v16h bb = load_b_frag_sw(Wf1, ks * 4 + nt, lane);
        acc = wmma16(a, bb, acc);
      }
#pragma unroll
      for (int i = 0; i < 8; ++i) {
        float v = acc[i] + b1c[nt];
        v = v > 0.f ? v : 0.f;
        v = v * g1c[nt] + t1c[nt];
        Hw[(i + 8 * half) * 64 + nt * 16 + nlo] = (_Float16)v;
      }
    }

    // --- layer 2: 16 x 64 @ 64 x 64, running max ---
#pragma unroll
    for (int nt = 0; nt < 4; ++nt) {
      v8f acc;
#pragma unroll
      for (int i = 0; i < 8; ++i) acc[i] = 0.f;
#pragma unroll
      for (int ks = 0; ks < 2; ++ks) {
        v16h a = load_a_frag(Hw, 64, ks * 32, lane);
        acc = wmma16(a, w2f[ks * 4 + nt], acc);
      }
#pragma unroll
      for (int i = 0; i < 8; ++i) {
        float v = acc[i] + b2c[nt];
        v = v > 0.f ? v : 0.f;
        v = v * g2c[nt] + t2c[nt];
        vmax[nt][i] = fmaxf(vmax[nt][i], v);
      }
    }
  }

  // --- cross-wave max (reuse A/H LDS region) ---
  __syncthreads();
  float* mw = maxbuf + w * 16 * 64;
#pragma unroll
  for (int nt = 0; nt < 4; ++nt)
#pragma unroll
    for (int i = 0; i < 8; ++i)
      mw[(i + 8 * half) * 64 + nt * 16 + nlo] = vmax[nt][i];
  __syncthreads();
  for (int o = tid; o < 16 * 64; o += 128) {
    float v = maxbuf[o];
    v = fmaxf(v, maxbuf[1024 + o]);
    v = fmaxf(v, maxbuf[2048 + o]);
    v = fmaxf(v, maxbuf[3072 + o]);
    y[(size_t)(p0 + (o >> 6)) * 64 + (o & 63)] = v;
  }
}

// ---------------- lin1 ([x1|x2|x3] 192 -> 1024) fused with global max ------
// grid = (NTOT/64, 16 col-groups); block = 128 thr (4 waves x 16 points).

__global__ __launch_bounds__(128) void lin1_kernel(
    const float* __restrict__ x1, const float* __restrict__ x2,
    const float* __restrict__ x3, const float* __restrict__ W,
    const float* __restrict__ b, const float* __restrict__ g,
    const float* __restrict__ t, unsigned* __restrict__ gkey) {
  __shared__ __align__(16) _Float16 Wf[24 * 512];      // 24 KB, swizzled
  __shared__ __align__(16) _Float16 Ast[4 * 16 * 192]; // 24 KB

  int tid = threadIdx.x;
  int lane = tid & 31, w = tid >> 5;
  int nlo = lane & 15, half = lane >> 4;
  int cg = blockIdx.y;

  for (int i = tid; i < 24 * 512; i += 128) {
    int k, n;
    sw_decode(i, k, n);
    Wf[i] = (_Float16)W[(size_t)k * 1024 + cg * 64 + n];
  }

  int p0 = blockIdx.x * 64 + w * 16;
  int p = p0 + nlo;
  _Float16* Aw = Ast + w * 16 * 192;
  for (int c0 = 0; c0 < 96; ++c0) {
    int c = half * 96 + c0;
    float v;
    if (c < 64)       v = x1[(size_t)p * 64 + c];
    else if (c < 128) v = x2[(size_t)p * 64 + (c - 64)];
    else              v = x3[(size_t)p * 64 + (c - 128)];
    Aw[nlo * 192 + c] = (_Float16)v;
  }
  __syncthreads();

  int batch = p0 >> 12;
#pragma unroll
  for (int nt = 0; nt < 4; ++nt) {
    v8f acc;
#pragma unroll
    for (int i = 0; i < 8; ++i) acc[i] = 0.f;
#pragma unroll
    for (int ks = 0; ks < 6; ++ks) {
      v16h a = load_a_frag(Aw, 192, ks * 32, lane);
      v16h bb = load_b_frag_sw(Wf, ks * 4 + nt, lane);
      acc = wmma16(a, bb, acc);
    }
    int col = cg * 64 + nt * 16 + nlo;
    float bc = b[col], gc = g[col] * BN_SCALE, tc = t[col];
    float mymax = -FLT_MAX;
#pragma unroll
    for (int i = 0; i < 8; ++i) {
      float v = acc[i] + bc;
      v = v > 0.f ? v : 0.f;
      v = v * gc + tc;
      mymax = fmaxf(mymax, v);
    }
    mymax = fmaxf(mymax, __shfl_xor(mymax, 16));
    if (half == 0) atomicMax(&gkey[batch * 1024 + col], f2key(mymax));
  }
}

__global__ __launch_bounds__(256) void init_gkey_kernel(unsigned* __restrict__ gkey) {
  int i = blockIdx.x * 256 + threadIdx.x;
  if (i < 16 * 1024) gkey[i] = 0u;  // below key of any finite float
}

// ---------------- head: mlp1(1024->512) -> mlp2(512->256) -> lin_out -------

__global__ __launch_bounds__(256) void head_kernel(
    const unsigned* __restrict__ gkey,
    const float* __restrict__ W1, const float* __restrict__ b1,
    const float* __restrict__ g1, const float* __restrict__ t1,
    const float* __restrict__ W2, const float* __restrict__ b2,
    const float* __restrict__ g2, const float* __restrict__ t2,
    const float* __restrict__ Wo, const float* __restrict__ bo,
    float* __restrict__ out) {
  __shared__ float gv[1024];
  __shared__ float h1[512];
  __shared__ float h2[256];
  int bb = blockIdx.x, t = threadIdx.x;

  for (int c = t; c < 1024; c += 256) gv[c] = key2f(gkey[bb * 1024 + c]);
  __syncthreads();
  for (int o = t; o < 512; o += 256) {
    float s = b1[o];
    for (int k = 0; k < 1024; ++k) s += gv[k] * W1[(size_t)k * 512 + o];
    s = s > 0.f ? s : 0.f;
    h1[o] = s * g1[o] * BN_SCALE + t1[o];
  }
  __syncthreads();
  if (t < 256) {
    float s = b2[t];
    for (int k = 0; k < 512; ++k) s += h1[k] * W2[(size_t)k * 256 + t];
    s = s > 0.f ? s : 0.f;
    h2[t] = s * g2[t] * BN_SCALE + t2[t];
  }
  __syncthreads();
  if (t < 256) {
    float s = bo[t];
    for (int k = 0; k < 256; ++k) s += h2[k] * Wo[(size_t)k * 256 + t];
    out[(size_t)bb * 256 + t] = s;
  }
}

// ---------------------------------------------------------------------------

extern "C" void kernel_launch(void* const* d_in, const int* in_sizes, int n_in,
                              void* d_out, int out_size, void* d_ws,
                              size_t ws_size, hipStream_t stream) {
  (void)in_sizes; (void)n_in; (void)out_size; (void)ws_size;
  // Input order (insertion order of setup_inputs / params):
  // 0: pos, 1: batch,
  // 2..9  conv1 (W,b,gamma,beta)x2 ; 10..17 conv2 ; 18..25 conv3
  // 26..29 lin1 (W,b,gamma,beta) ; 30..33 mlp1 ; 34..37 mlp2 ; 38..39 lin_out
  const float* pos = (const float*)d_in[0];
  auto F = [&](int i) { return (const float*)d_in[i]; };

  char* ws = (char*)d_ws;
  auto alloc = [&](size_t bytes) {
    char* p = ws;
    ws += (bytes + 255) & ~size_t(255);
    return p;
  };
  float* x1 = (float*)alloc((size_t)NTOT * 64 * 4);
  float* x2 = (float*)alloc((size_t)NTOT * 64 * 4);
  float* x3 = (float*)alloc((size_t)NTOT * 64 * 4);
  int*   ki = (int*)alloc((size_t)NTOT * KNN * 4);
  float* xn = (float*)alloc((size_t)NTOT * 4);
  unsigned* gk = (unsigned*)alloc((size_t)16 * 1024 * 4);

  // ---- conv1: input dim 3 ----
  norm_kernel<<<NTOT / 256, 256, 0, stream>>>(pos, 3, xn);
  knn_kernel<<<NTOT, 256, 0, stream>>>(pos, xn, 3, ki);
  edgeconv_kernel<32, 3><<<NTOT / 16, 128, 0, stream>>>(
      pos, ki, F(2), F(3), F(4), F(5), F(6), F(7), F(8), F(9), x1);

  // ---- conv2 ----
  norm_kernel<<<NTOT / 256, 256, 0, stream>>>(x1, 64, xn);
  knn_kernel<<<NTOT, 256, 0, stream>>>(x1, xn, 64, ki);
  edgeconv_kernel<128, 64><<<NTOT / 16, 128, 0, stream>>>(
      x1, ki, F(10), F(11), F(12), F(13), F(14), F(15), F(16), F(17), x2);

  // ---- conv3 ----
  norm_kernel<<<NTOT / 256, 256, 0, stream>>>(x2, 64, xn);
  knn_kernel<<<NTOT, 256, 0, stream>>>(x2, xn, 64, ki);
  edgeconv_kernel<128, 64><<<NTOT / 16, 128, 0, stream>>>(
      x2, ki, F(18), F(19), F(20), F(21), F(22), F(23), F(24), F(25), x3);

  // ---- lin1 + global max pool ----
  init_gkey_kernel<<<64, 256, 0, stream>>>(gk);
  dim3 g1(NTOT / 64, 16);
  lin1_kernel<<<g1, 128, 0, stream>>>(x1, x2, x3, F(26), F(27), F(28), F(29), gk);

  // ---- head ----
  head_kernel<<<16, 256, 0, stream>>>(gk, F(30), F(31), F(32), F(33), F(34),
                                      F(35), F(36), F(37), F(38), F(39),
                                      (float*)d_out);
}
